// EdgeGATLayer_43533788512637
// MI455X (gfx1250) — compile-verified, matrix operand order checked by reading, not run on hardware
//
#include <hip/hip_runtime.h>
#include <cstdint>
#include <cstddef>

// ---------------------------------------------------------------------------
// EdgeGAT layer for MI455X (gfx1250, wave32, WMMA).
// GEMMs run on v_wmma_f32_16x16x32_f16 (f32 accumulate); everything else is
// memory-bound streaming + L2 atomics.
// ---------------------------------------------------------------------------

#define NNODES 50000
#define NEDGES 800000
#define DD     64
#define KIN    192        // 3*D concat input width

typedef _Float16 v16h __attribute__((ext_vector_type(16)));
typedef float    v8f  __attribute__((ext_vector_type(8)));
typedef _Float16 h8   __attribute__((ext_vector_type(8)));

// ----------------------------- helpers -------------------------------------

__device__ __forceinline__ float gelu_tanh(float x) {
    // jax.nn.gelu approximate=True
    float x3 = x * x * x;
    float t  = tanhf(0.7978845608028654f * (x + 0.044715f * x3));
    return 0.5f * x * (1.0f + t);
}

// order-preserving float->uint encoding for atomicMax on floats
__device__ __forceinline__ unsigned enc_f32(float f) {
    unsigned u = __float_as_uint(f);
    return (u & 0x80000000u) ? ~u : (u | 0x80000000u);
}
__device__ __forceinline__ float dec_f32(unsigned k) {
    unsigned u = (k & 0x80000000u) ? (k & 0x7FFFFFFFu) : ~k;
    return __uint_as_float(u);
}

// A-fragment for V_WMMA_F32_16X16X32_F16 (16x32 f16, MxK):
//   lane L holds row M = L&15; K-halfgroup base = (L>>4)*8
//   elements 0..7  <- K = kb+0..7   (VGPR0..3)
//   elements 8..15 <- K = kb+16..23 (VGPR4..7)
// segp points at the start of this row's 32-wide K block (f32 source data).
__device__ __forceinline__ v16h make_a_frag(const float* segp, int hi) {
    const float4* p0 = (const float4*)(segp + hi * 8);
    const float4* p1 = (const float4*)(segp + hi * 8 + 16);
    float4 a0 = p0[0], a1 = p0[1];
    float4 b0 = p1[0], b1 = p1[1];
    v16h f;
    f[0]  = (_Float16)a0.x; f[1]  = (_Float16)a0.y;
    f[2]  = (_Float16)a0.z; f[3]  = (_Float16)a0.w;
    f[4]  = (_Float16)a1.x; f[5]  = (_Float16)a1.y;
    f[6]  = (_Float16)a1.z; f[7]  = (_Float16)a1.w;
    f[8]  = (_Float16)b0.x; f[9]  = (_Float16)b0.y;
    f[10] = (_Float16)b0.z; f[11] = (_Float16)b0.w;
    f[12] = (_Float16)b1.x; f[13] = (_Float16)b1.y;
    f[14] = (_Float16)b1.z; f[15] = (_Float16)b1.w;
    return f;
}

// B-fragment (32x16 f16, KxN), mirrored layout: lane holds column N = L&15,
// same K striping as A.  wrow = &Wh[n*KIN], k0 = kk*32 + (lane>>4)*8.
__device__ __forceinline__ v16h make_b_frag(const _Float16* wrow, int k0) {
    h8 lo = *(const h8*)(wrow + k0);
    h8 up = *(const h8*)(wrow + k0 + 16);
    v16h f;
#pragma unroll
    for (int i = 0; i < 8; ++i) { f[i] = lo[i]; f[i + 8] = up[i]; }
    return f;
}

struct AccTile { v8f a0, a1, a2, a3; };

// One wave computes a [16 rows x 64 cols] GEMM tile:
//   out = concat(s0row, s1row, s2row)[16,192] @ Wh^T[192,64]
// s0/s1/s2 are the three 64-float row pointers for THIS lane's row (lane&15).
__device__ __forceinline__ AccTile gemm_tile(const float* s0, const float* s1,
                                             const float* s2,
                                             const _Float16* __restrict__ Wh,
                                             int lane) {
    const int ncol = lane & 15;
    const int hi   = lane >> 4;
    const _Float16* w0 = Wh + (size_t)(ncol)      * KIN;
    const _Float16* w1 = Wh + (size_t)(16 + ncol) * KIN;
    const _Float16* w2 = Wh + (size_t)(32 + ncol) * KIN;
    const _Float16* w3 = Wh + (size_t)(48 + ncol) * KIN;
    const float* segs[3] = { s0, s1, s2 };

    AccTile t;
    t.a0 = v8f{}; t.a1 = v8f{}; t.a2 = v8f{}; t.a3 = v8f{};
#pragma unroll
    for (int kk = 0; kk < 6; ++kk) {
        const float* sp = segs[kk >> 1] + (kk & 1) * 32;
        v16h a  = make_a_frag(sp, hi);
        int  k0 = kk * 32 + hi * 8;
        t.a0 = __builtin_amdgcn_wmma_f32_16x16x32_f16(false, a, false,
                 make_b_frag(w0, k0), (short)0, t.a0, false, false);
        t.a1 = __builtin_amdgcn_wmma_f32_16x16x32_f16(false, a, false,
                 make_b_frag(w1, k0), (short)0, t.a1, false, false);
        t.a2 = __builtin_amdgcn_wmma_f32_16x16x32_f16(false, a, false,
                 make_b_frag(w2, k0), (short)0, t.a2, false, false);
        t.a3 = __builtin_amdgcn_wmma_f32_16x16x32_f16(false, a, false,
                 make_b_frag(w3, k0), (short)0, t.a3, false, false);
    }
    return t;
}

// ----------------------------- kernels -------------------------------------

// K0: f32 -> f16 weight conversion (both matrices, 64*192 each)
__global__ void k_convert_w(const float* __restrict__ eW,
                            const float* __restrict__ nW,
                            _Float16* __restrict__ eWh,
                            _Float16* __restrict__ nWh) {
    int i = blockIdx.x * blockDim.x + threadIdx.x;
    if (i < DD * KIN) {
        eWh[i] = (_Float16)eW[i];
        nWh[i] = (_Float16)nW[i];
    }
}

// K1: per-edge attention logit + segment max (one wave per edge)
__global__ void k_logits(const float* __restrict__ nodes,
                         const int* __restrict__ senders,
                         const int* __restrict__ receivers,
                         float* __restrict__ logits,
                         unsigned* __restrict__ segmax) {
    int lane = threadIdx.x & 31;
    int e = blockIdx.x * (blockDim.x >> 5) + (threadIdx.x >> 5);
    if (e >= NEDGES) return;
    int s = senders[e], r = receivers[e];
    const float* ps = nodes + (size_t)s * DD;
    const float* pr = nodes + (size_t)r * DD;
    float acc = ps[lane] * pr[lane] + ps[lane + 32] * pr[lane + 32];
#pragma unroll
    for (int off = 16; off >= 1; off >>= 1) acc += __shfl_xor(acc, off, 32);
    if (lane == 0) {
        float lg = acc * 0.125f;          // 1/sqrt(64)
        logits[e] = lg;
        atomicMax(segmax + r, enc_f32(lg));
    }
}

// K2: per-edge exp(logit - segmax) + segment sum
__global__ void k_expsum(const float* __restrict__ logits,
                         const int* __restrict__ receivers,
                         const unsigned* __restrict__ segmax,
                         float* __restrict__ ex,
                         float* __restrict__ segsum) {
    int e = blockIdx.x * blockDim.x + threadIdx.x;
    if (e >= NEDGES) return;
    int r = receivers[e];
    float v = expf(logits[e] - dec_f32(segmax[r]));
    ex[e] = v;
    atomicAdd(segsum + r, v);
}

// K3: edge update GEMM + GELU + attention weighting + scatter-add
__global__ void k_edge_update(const float* __restrict__ edges,
                              const float* __restrict__ nodes,
                              const int* __restrict__ senders,
                              const int* __restrict__ receivers,
                              const _Float16* __restrict__ eWh,
                              const float* __restrict__ ebias,
                              const float* __restrict__ ex,
                              const float* __restrict__ segsum,
                              float* __restrict__ e_w_out,
                              float* __restrict__ outgoing,
                              float* __restrict__ incoming) {
    int lane = threadIdx.x & 31;
    int tile = blockIdx.x * (blockDim.x >> 5) + (threadIdx.x >> 5);
    if (tile * 16 >= NEDGES) return;          // wave-uniform

    int row = lane & 15;
    int hi  = lane >> 4;
    int e   = tile * 16 + row;
    int s   = senders[e];
    int r   = receivers[e];
    float att = ex[e] / segsum[r];            // identical in lane & lane+16

    AccTile t = gemm_tile(edges + (size_t)e * DD,
                          nodes + (size_t)s * DD,
                          nodes + (size_t)r * DD, eWh, lane);

    int col = row;                            // C/D layout: N = lane&15
    v8f accs[4] = { t.a0, t.a1, t.a2, t.a3 };
#pragma unroll
    for (int nt = 0; nt < 4; ++nt) {
        float b = ebias[nt * 16 + col];
#pragma unroll
        for (int m = 0; m < 8; ++m) {
            int   rowm = m + 8 * hi;          // C/D layout: M = vgpr + 8*(lane>=16)
            float g    = gelu_tanh(accs[nt][m] + b);
            float v    = g * __shfl(att, rowm, 32);
            int   sm   = __shfl(s, rowm, 32);
            int   rm   = __shfl(r, rowm, 32);
            int   em   = tile * 16 + rowm;
            e_w_out[(size_t)em * DD + nt * 16 + col] = v;
            atomicAdd(outgoing + (size_t)sm * DD + nt * 16 + col, v);
            atomicAdd(incoming + (size_t)rm * DD + nt * 16 + col, v);
        }
    }
}

// K4: node update GEMM + GELU
__global__ void k_node_update(const float* __restrict__ nodes,
                              const float* __restrict__ outgoing,
                              const float* __restrict__ incoming,
                              const _Float16* __restrict__ nWh,
                              const float* __restrict__ nbias,
                              float* __restrict__ n_out) {
    int lane = threadIdx.x & 31;
    int tile = blockIdx.x * (blockDim.x >> 5) + (threadIdx.x >> 5);
    if (tile * 16 >= NNODES) return;          // wave-uniform

    int row = lane & 15;
    int hi  = lane >> 4;
    int nid = tile * 16 + row;

    AccTile t = gemm_tile(nodes    + (size_t)nid * DD,
                          outgoing + (size_t)nid * DD,
                          incoming + (size_t)nid * DD, nWh, lane);

    int col = row;
    v8f accs[4] = { t.a0, t.a1, t.a2, t.a3 };
#pragma unroll
    for (int nt = 0; nt < 4; ++nt) {
        float b = nbias[nt * 16 + col];
#pragma unroll
        for (int m = 0; m < 8; ++m) {
            int rowm = m + 8 * hi;
            int nm   = tile * 16 + rowm;
            float g  = gelu_tanh(accs[nt][m] + b);
            n_out[(size_t)nm * DD + nt * 16 + col] = g;
        }
    }
}

// ----------------------------- launch --------------------------------------

extern "C" void kernel_launch(void* const* d_in, const int* in_sizes, int n_in,
                              void* d_out, int out_size, void* d_ws,
                              size_t ws_size, hipStream_t stream) {
    const float* nodes     = (const float*)d_in[0];
    const float* edges     = (const float*)d_in[1];
    const int*   senders   = (const int*)d_in[2];
    const int*   receivers = (const int*)d_in[3];
    const float* edge_W    = (const float*)d_in[4];
    const float* edge_b    = (const float*)d_in[5];
    const float* node_W    = (const float*)d_in[6];
    const float* node_b    = (const float*)d_in[7];

    float* n_upd = (float*)d_out;                          // [N, D]
    float* e_w   = (float*)d_out + (size_t)NNODES * DD;    // [E, D]

    // workspace layout (bytes)
    char* ws = (char*)d_ws;
    const size_t OFF_EWH    = 0;                       // 64*192*2 = 24576
    const size_t OFF_NWH    = 24576;                   // 24576
    const size_t OFF_LOGITS = 49152;                   // E*4 = 3,200,000
    const size_t OFF_EX     = OFF_LOGITS + 3200000;    // 3,200,000
    const size_t OFF_SEGMAX = OFF_EX + 3200000;        // N*4 = 200,000  (zeroed)
    const size_t OFF_SEGSUM = OFF_SEGMAX + 200000;     // N*4 = 200,000  (zeroed)
    const size_t OFF_OUTG   = OFF_SEGSUM + 200000;     // N*D*4 = 12,800,000 (zeroed)
    const size_t OFF_INC    = OFF_OUTG + 12800000;     // 12,800,000 (zeroed)
    const size_t WS_NEED    = OFF_INC + 12800000;
    if (ws_size < WS_NEED) return;

    _Float16* eWh    = (_Float16*)(ws + OFF_EWH);
    _Float16* nWh    = (_Float16*)(ws + OFF_NWH);
    float*    logits = (float*)(ws + OFF_LOGITS);
    float*    ex     = (float*)(ws + OFF_EX);
    unsigned* segmax = (unsigned*)(ws + OFF_SEGMAX);
    float*    segsum = (float*)(ws + OFF_SEGSUM);
    float*    outg   = (float*)(ws + OFF_OUTG);
    float*    inc    = (float*)(ws + OFF_INC);

    // zero the accumulated region (segmax key 0 == below every finite float)
    hipMemsetAsync(ws + OFF_SEGMAX, 0, WS_NEED - OFF_SEGMAX, stream);

    // K0: weight conversion
    k_convert_w<<<(DD * KIN + 255) / 256, 256, 0, stream>>>(edge_W, node_W,
                                                            eWh, nWh);
    // K1: logits + segment max (8 waves/block, 1 wave per edge)
    k_logits<<<NEDGES / 8, 256, 0, stream>>>(nodes, senders, receivers, logits,
                                             segmax);
    // K2: exp + segment sum
    k_expsum<<<(NEDGES + 255) / 256, 256, 0, stream>>>(logits, receivers,
                                                       segmax, ex, segsum);
    // K3: edge GEMM (+GELU, attention, scatter) — 50000 tiles, 8 waves/block
    k_edge_update<<<(NEDGES / 16) / 8, 256, 0, stream>>>(
        edges, nodes, senders, receivers, eWh, edge_b, ex, segsum, e_w, outg,
        inc);
    // K4: node GEMM (+GELU) — 3125 tiles
    k_node_update<<<((NNODES / 16) + 7) / 8, 256, 0, stream>>>(
        nodes, outg, inc, nWh, node_b, n_upd);
}
